// AttnBlock_67757404061729
// MI455X (gfx1250) — matrix operand order repaired and checked by reading, not compiled, not run on hardware
//
#include <hip/hip_runtime.h>

// Problem constants (match reference): B=2, C=512, G=32, H=W=64, N=4096.
#define C_DIM 512
#define N_POS 4096

typedef __attribute__((ext_vector_type(16))) _Float16 v16h;
typedef __attribute__((ext_vector_type(8)))  _Float16 v8h;
typedef __attribute__((ext_vector_type(8)))  float    v8f;

__device__ __forceinline__ v8f wmma16(v16h a, v16h b, v8f c) {
  // D = A(16x32 f16) * B(32x16 f16) + C(16x16 f32)
  return __builtin_amdgcn_wmma_f32_16x16x32_f16(false, a, false, b, (short)0, c,
                                                false, false);
}

// --- WMMA operand loaders (contiguous k = 16*(lane/16) + s, wave32) ---------------
// A tile 16(M) x 32(K), f16, from row-major buffer [M][ld].
__device__ __forceinline__ v16h load_a(const _Float16* base, int ld, int lane) {
  const int row = lane & 15, kh = lane >> 4;
  const _Float16* p = base + row * ld + kh * 16;
  v16h a = *(const v16h*)(p);
  return a;
}

// B tile 32(K) x 16(N), f16, from an N-major buffer: element (k,n) at base[n*ld + k]:
// per lane one contiguous 32B read.
__device__ __forceinline__ v16h load_bt(const _Float16* base, int ld, int lane) {
  const int n = lane & 15, kh = lane >> 4;
  return *(const v16h*)(base + (size_t)n * ld + kh * 16);
}

// --- 1) GroupNorm: x[B,C,H,W] -> Hbuf[B][N][C] (f16, position-major) -------
__global__ __launch_bounds__(256) void gn_kernel(const float* __restrict__ x,
                                                 const float* __restrict__ gw,
                                                 const float* __restrict__ gb,
                                                 _Float16* __restrict__ Hbuf) {
  const int CPG = 16;                       // C/G
  const int bg = blockIdx.x;                // B*G = 64 blocks
  const int b = bg >> 5, g = bg & 31;
  const float* xp = x + ((size_t)b * C_DIM + g * CPG) * N_POS;
  const int t = threadIdx.x;

  float s = 0.f, ss = 0.f;
  for (int i = t; i < CPG * N_POS; i += 256) { float v = xp[i]; s += v; ss += v * v; }
  __shared__ float rs[256], rss[256];
  rs[t] = s; rss[t] = ss;
  __syncthreads();
  for (int off = 128; off; off >>= 1) {
    if (t < off) { rs[t] += rs[t + off]; rss[t] += rss[t + off]; }
    __syncthreads();
  }
  const float inv_n = 1.f / (float)(CPG * N_POS);
  const float mu = rs[0] * inv_n;
  const float var = rss[0] * inv_n - mu * mu;
  const float rstd = rsqrtf(var + 1e-6f);

  __shared__ float swsc[16], sbsc[16];
  if (t < 16) {
    const int c = g * CPG + t;
    const float wsc = gw[c] * rstd;
    swsc[t] = wsc;
    sbsc[t] = gb[c] - mu * wsc;
  }
  __syncthreads();

  // one 32B store (16 channels of this group) per position
  for (int n = t; n < N_POS; n += 256) {
    v16h hv;
#pragma unroll
    for (int cc = 0; cc < 16; ++cc)
      hv[cc] = (_Float16)(xp[(size_t)cc * N_POS + n] * swsc[cc] + sbsc[cc]);
    *(v16h*)(Hbuf + ((size_t)b * N_POS + n) * C_DIM + g * CPG) = hv;
  }
}

// --- 2) Convert the four weight matrices to f16 ----------------------------
__global__ __launch_bounds__(256) void cvtw_kernel(const float* w0, const float* w1,
                                                   const float* w2, const float* w3,
                                                   _Float16* o0, _Float16* o1,
                                                   _Float16* o2, _Float16* o3) {
  const int i = blockIdx.x * 256 + threadIdx.x;   // gridDim.x = 1024 -> 262144
  const float* s; _Float16* d;
  switch (blockIdx.y) {
    case 0: s = w0; d = o0; break;
    case 1: s = w1; d = o1; break;
    case 2: s = w2; d = o2; break;
    default: s = w3; d = o3; break;
  }
  d[i] = (_Float16)s[i];
}

// --- 3) QKV projection: D[n,d] = sum_c H[n,c] * W[d,c] + bias --------------
// grid: (N/16, B, 3); 256 threads = 8 waves.
// kk-outer, 4 resident accumulators, depth-1 software pipeline on B tiles.
__global__ __launch_bounds__(256) void qkv_kernel(
    const _Float16* __restrict__ Hbuf, const _Float16* __restrict__ Wq,
    const _Float16* __restrict__ Wk, const _Float16* __restrict__ Wv,
    const float* __restrict__ bq, const float* __restrict__ bk,
    const float* __restrict__ bv, _Float16* __restrict__ Qbuf,
    _Float16* __restrict__ Kbuf, _Float16* __restrict__ Vbuf) {
  const int n0 = blockIdx.x * 16, b = blockIdx.y, which = blockIdx.z;
  __shared__ __align__(32) _Float16 sA[16 * C_DIM];      // 16 rows of H, 16KB
  __shared__ _Float16 sT[8][16 * 17];                    // per-wave transpose buf
  const int t = threadIdx.x, wave = t >> 5, lane = t & 31;

  const _Float16* hsrc = Hbuf + ((size_t)b * N_POS + n0) * C_DIM;  // contiguous 16KB
  for (int i = t * 8; i < 16 * C_DIM; i += 256 * 8)
    *(v8h*)(sA + i) = *(const v8h*)(hsrc + i);
  __syncthreads();

  const _Float16* W = (which == 0) ? Wq : (which == 1) ? Wk : Wv;
  const float* bias = (which == 0) ? bq : (which == 1) ? bk : bv;
  // fold scale*log2(e) into Q so softmax can use exp2
  const float qs = 0.044194173824159216f * 1.4426950408889634f;

  v8f acc[4] = {{}, {}, {}, {}};
  const _Float16* Wb = W + (size_t)(wave * 16) * C_DIM;  // d-tile stream base
  v16h bcur[4], bnxt[4];
#pragma unroll
  for (int i = 0; i < 4; ++i)
    bcur[i] = load_bt(Wb + (size_t)i * 128 * C_DIM, C_DIM, lane);
#pragma unroll
  for (int kk = 0; kk < 16; ++kk) {
    if (kk < 15) {
#pragma unroll
      for (int i = 0; i < 4; ++i)
        bnxt[i] = load_bt(Wb + (size_t)i * 128 * C_DIM + (kk + 1) * 32, C_DIM, lane);
    }
    v16h a = load_a(sA + kk * 32, C_DIM, lane);
#pragma unroll
    for (int i = 0; i < 4; ++i) acc[i] = wmma16(a, bcur[i], acc[i]);
#pragma unroll
    for (int i = 0; i < 4; ++i) bcur[i] = bnxt[i];
  }

  const int col = lane & 15, rb = (lane >> 4) * 8;
#pragma unroll
  for (int i = 0; i < 4; ++i) {
    const int d0 = (wave + 8 * i) * 16;
    const int d = d0 + col;
    const float bb = bias[d];
    if (which == 0) {
#pragma unroll
      for (int v = 0; v < 8; ++v)
        Qbuf[((size_t)b * N_POS + n0 + rb + v) * C_DIM + d] =
            (_Float16)((acc[i][v] + bb) * qs);
    } else if (which == 1) {
#pragma unroll
      for (int v = 0; v < 8; ++v)
        Kbuf[((size_t)b * N_POS + n0 + rb + v) * C_DIM + d] =
            (_Float16)(acc[i][v] + bb);
    } else {
      // V stored channel-major [C][N]: transpose the 16x16 tile through LDS
      _Float16* tp = sT[wave];
#pragma unroll
      for (int v = 0; v < 8; ++v)
        tp[(rb + v) * 17 + col] = (_Float16)(acc[i][v] + bb);
      // same-wave LDS ops are in order; read back transposed
      const int dd = lane & 15, h = lane >> 4;
      v8h ov;
#pragma unroll
      for (int u = 0; u < 8; ++u) ov[u] = tp[(h * 8 + u) * 17 + dd];
      *(v8h*)(Vbuf + ((size_t)b * C_DIM + d0 + dd) * N_POS + n0 + h * 8) = ov;
    }
  }
}

// --- 4) Flash attention: 32-row query tile per workgroup -------------------
// grid (N/32, B); 256 threads = 8 waves. Wave w: score cols w*16.., O channels
// w*64... Each K/V B-tile feeds TWO WMMAs (row-tiles 0 and 1): halves L2 traffic.
__global__ __launch_bounds__(256) void attn_kernel(const _Float16* __restrict__ Qbuf,
                                                   const _Float16* __restrict__ Kbuf,
                                                   const _Float16* __restrict__ Vbuf,
                                                   _Float16* __restrict__ Obuf) {
  const int KT = 128;  // key-tile
  const int i0 = blockIdx.x * 32, b = blockIdx.y;
  __shared__ __align__(32) _Float16 sQ[32 * C_DIM];   // 32KB
  __shared__ __align__(16) float    sS[32 * KT];      // 16KB
  __shared__ __align__(32) _Float16 sP[32 * KT];      // 8KB
  __shared__ float rowM[32], rowL[32], rowF[32];

  const int t = threadIdx.x, wave = t >> 5, lane = t & 31;
  const int col = lane & 15, rb = (lane >> 4) * 8;
  const int cbase = wave * 64;

  const _Float16* qsrc = Qbuf + ((size_t)b * N_POS + i0) * C_DIM;
  for (int i = t * 8; i < 32 * C_DIM; i += 256 * 8)
    *(v8h*)(sQ + i) = *(const v8h*)(qsrc + i);
  if (t < 32) { rowM[t] = -1e30f; rowL[t] = 0.f; }
  __syncthreads();

  v8f oo0[4] = {{}, {}, {}, {}};  // rows 0-15  x 64 channels
  v8f oo1[4] = {{}, {}, {}, {}};  // rows 16-31 x 64 channels

  for (int j0 = 0; j0 < N_POS; j0 += KT) {
    // prefetch next iteration's K tile and V slice (gfx1250 global_prefetch_b8)
    if (j0 + KT < N_POS) {
      const char* nk = (const char*)(Kbuf +
          ((size_t)b * N_POS + j0 + KT + wave * 16) * C_DIM);
#pragma unroll
      for (int p = 0; p < 8; ++p)
        __builtin_prefetch(nk + p * 2048 + lane * 64, 0, 1);
      const char* nv = (const char*)(Vbuf +
          ((size_t)b * C_DIM + cbase) * N_POS + j0 + KT);
#pragma unroll
      for (int p = 0; p < 2; ++p)
        __builtin_prefetch(nv + ((size_t)(p * 32) + lane) * (N_POS * 2), 0, 1);
    }

    // ---- stage 1: scores S[32, wave*16..+15]; K-reduction over 512 channels.
    // depth-2 pipeline on K B-tiles; each B-tile feeds both row-tile WMMAs.
    v8f s0 = {}, s1 = {};
    const _Float16* kbase = Kbuf + ((size_t)b * N_POS + j0 + wave * 16) * C_DIM;
    v16h kb0 = load_bt(kbase, C_DIM, lane);
    v16h kb1 = load_bt(kbase + 32, C_DIM, lane);
#pragma unroll
    for (int kk = 0; kk < 16; kk += 2) {
      v16h kbn0, kbn1;
      if (kk < 14) {
        kbn0 = load_bt(kbase + (kk + 2) * 32, C_DIM, lane);
        kbn1 = load_bt(kbase + (kk + 3) * 32, C_DIM, lane);
      }
      v16h a00 = load_a(sQ + kk * 32, C_DIM, lane);
      v16h a10 = load_a(sQ + 16 * C_DIM + kk * 32, C_DIM, lane);
      s0 = wmma16(a00, kb0, s0);
      s1 = wmma16(a10, kb0, s1);
      v16h a01 = load_a(sQ + (kk + 1) * 32, C_DIM, lane);
      v16h a11 = load_a(sQ + 16 * C_DIM + (kk + 1) * 32, C_DIM, lane);
      s0 = wmma16(a01, kb1, s0);
      s1 = wmma16(a11, kb1, s1);
      kb0 = kbn0;
      kb1 = kbn1;
    }
#pragma unroll
    for (int v = 0; v < 8; ++v) {
      sS[(rb + v) * KT + wave * 16 + col] = s0[v];
      sS[(16 + rb + v) * KT + wave * 16 + col] = s1[v];
    }
    __syncthreads();

    // ---- stage 2: online softmax. thread t: row t/8, 16 cols.
    {
      const int row = t >> 3, c0 = (t & 7) * 16;
      float* sr = sS + row * KT + c0;
      float lm = -1e30f;
#pragma unroll
      for (int k2 = 0; k2 < 16; ++k2) lm = fmaxf(lm, sr[k2]);
#pragma unroll
      for (int off = 1; off < 8; off <<= 1) lm = fmaxf(lm, __shfl_xor(lm, off, 8));
      const float mo = rowM[row];
      const float mn = fmaxf(mo, lm);
      const float fac = exp2f(mo - mn);
      float ls = 0.f;
      _Float16* pr = sP + row * KT + c0;
#pragma unroll
      for (int k2 = 0; k2 < 16; ++k2) {
        float p = exp2f(sr[k2] - mn);
        ls += p;
        pr[k2] = (_Float16)p;
      }
#pragma unroll
      for (int off = 1; off < 8; off <<= 1) ls += __shfl_xor(ls, off, 8);
      if ((t & 7) == 0) { rowM[row] = mn; rowL[row] = rowL[row] * fac + ls; rowF[row] = fac; }
    }
    __syncthreads();

    // ---- stage 3: rescale O, then O += P[32,128] @ Vt[128, 64-slice];
    // each V B-tile feeds both row-tile WMMAs.
    {
      float f0[8], f1[8];
#pragma unroll
      for (int v = 0; v < 8; ++v) { f0[v] = rowF[rb + v]; f1[v] = rowF[16 + rb + v]; }
#pragma unroll
      for (int i = 0; i < 4; ++i)
#pragma unroll
        for (int v = 0; v < 8; ++v) { oo0[i][v] *= f0[v]; oo1[i][v] *= f1[v]; }

      const _Float16* vb = Vbuf + ((size_t)b * C_DIM + cbase) * N_POS + j0;
#pragma unroll
      for (int kk = 0; kk < 4; ++kk) {
        v16h a0 = load_a(sP + kk * 32, KT, lane);
        v16h a1 = load_a(sP + 16 * KT + kk * 32, KT, lane);
#pragma unroll
        for (int i = 0; i < 4; ++i) {
          v16h vt = load_bt(vb + (size_t)i * 16 * N_POS + kk * 32, N_POS, lane);
          oo0[i] = wmma16(a0, vt, oo0[i]);
          oo1[i] = wmma16(a1, vt, oo1[i]);
        }
      }
    }
    // next iteration's post-stage-1 barrier orders sP/rowF reuse
  }

  // ---- epilogue: normalize rows, store O[i, c] position-major f16
  float invl0[8], invl1[8];
#pragma unroll
  for (int v = 0; v < 8; ++v) {
    invl0[v] = 1.f / rowL[rb + v];
    invl1[v] = 1.f / rowL[16 + rb + v];
  }
#pragma unroll
  for (int v = 0; v < 8; ++v) {
    const size_t b0 = ((size_t)b * N_POS + i0 + rb + v) * C_DIM + cbase + col;
    const size_t b1 = ((size_t)b * N_POS + i0 + 16 + rb + v) * C_DIM + cbase + col;
#pragma unroll
    for (int i = 0; i < 4; ++i) {
      Obuf[b0 + 16 * i] = (_Float16)(oo0[i][v] * invl0[v]);
      Obuf[b1 + 16 * i] = (_Float16)(oo1[i][v] * invl1[v]);
    }
  }
}

// --- 5) Output projection + residual: out = x + Wp*O + bp -----------------
__global__ __launch_bounds__(256) void proj_kernel(const _Float16* __restrict__ Obuf,
                                                   const _Float16* __restrict__ Wp,
                                                   const float* __restrict__ bp,
                                                   const float* __restrict__ x,
                                                   float* __restrict__ out) {
  const int n0 = blockIdx.x * 16, b = blockIdx.y;
  __shared__ __align__(32) _Float16 sA[16 * C_DIM];
  __shared__ float sTf[8][16 * 17];   // per-wave f32 transpose buffer
  const int t = threadIdx.x, wave = t >> 5, lane = t & 31;

  const _Float16* osrc = Obuf + ((size_t)b * N_POS + n0) * C_DIM;
  for (int i = t * 8; i < 16 * C_DIM; i += 256 * 8)
    *(v8h*)(sA + i) = *(const v8h*)(osrc + i);
  __syncthreads();

  v8f acc[4] = {{}, {}, {}, {}};
  const _Float16* Wb = Wp + (size_t)(wave * 16) * C_DIM;
  v16h bcur[4], bnxt[4];
#pragma unroll
  for (int i = 0; i < 4; ++i)
    bcur[i] = load_bt(Wb + (size_t)i * 128 * C_DIM, C_DIM, lane);
#pragma unroll
  for (int kk = 0; kk < 16; ++kk) {
    if (kk < 15) {
#pragma unroll
      for (int i = 0; i < 4; ++i)
        bnxt[i] = load_bt(Wb + (size_t)i * 128 * C_DIM + (kk + 1) * 32, C_DIM, lane);
    }
    v16h a = load_a(sA + kk * 32, C_DIM, lane);
#pragma unroll
    for (int i = 0; i < 4; ++i) acc[i] = wmma16(a, bcur[i], acc[i]);
#pragma unroll
    for (int i = 0; i < 4; ++i) bcur[i] = bnxt[i];
  }

  // transposed epilogue: residual-add and store along channel rows (32B chunks)
  const int col = lane & 15, rb = (lane >> 4) * 8;
  const int dd = lane & 15, h = lane >> 4;
  float* tp = sTf[wave];
#pragma unroll
  for (int i = 0; i < 4; ++i) {
    const int d0 = (wave + 8 * i) * 16;
#pragma unroll
    for (int v = 0; v < 8; ++v) tp[(rb + v) * 17 + col] = acc[i][v];
    // same-wave LDS ops are in order
    const float bb = bp[d0 + dd];
    const size_t ridx = ((size_t)b * C_DIM + d0 + dd) * N_POS + n0 + h * 8;
#pragma unroll
    for (int u = 0; u < 8; ++u)
      out[ridx + u] = x[ridx + u] + tp[(h * 8 + u) * 17 + dd] + bb;
  }
}

extern "C" void kernel_launch(void* const* d_in, const int* in_sizes, int n_in,
                              void* d_out, int out_size, void* d_ws, size_t ws_size,
                              hipStream_t stream) {
  (void)in_sizes; (void)n_in; (void)out_size; (void)ws_size;
  const float* x    = (const float*)d_in[0];
  const float* gn_w = (const float*)d_in[1];
  const float* gn_b = (const float*)d_in[2];
  const float* wq   = (const float*)d_in[3];
  const float* bq   = (const float*)d_in[4];
  const float* wk   = (const float*)d_in[5];
  const float* bk   = (const float*)d_in[6];
  const float* wv   = (const float*)d_in[7];
  const float* bv   = (const float*)d_in[8];
  const float* wp   = (const float*)d_in[9];
  const float* bp   = (const float*)d_in[10];
  float* out = (float*)d_out;

  // workspace layout (f16 buffers), ~42 MB total
  const size_t BNC = (size_t)2 * N_POS * C_DIM;           // 4M elems, 8MB f16
  const size_t WSZ = (size_t)C_DIM * C_DIM;               // 256K elems, 0.5MB f16
  char* ws = (char*)d_ws;
  _Float16* Hbuf = (_Float16*)(ws);                       // [B][N][C]
  _Float16* Qbuf = Hbuf + BNC;                            // [B][N][C]
  _Float16* Kbuf = Qbuf + BNC;                            // [B][N][C]
  _Float16* Vbuf = Kbuf + BNC;                            // [B][C][N]
  _Float16* Obuf = Vbuf + BNC;                            // [B][N][C]
  _Float16* Wqh  = Obuf + BNC;
  _Float16* Wkh  = Wqh + WSZ;
  _Float16* Wvh  = Wkh + WSZ;
  _Float16* Wph  = Wvh + WSZ;

  gn_kernel<<<64, 256, 0, stream>>>(x, gn_w, gn_b, Hbuf);
  cvtw_kernel<<<dim3(1024, 4), 256, 0, stream>>>(wq, wk, wv, wp, Wqh, Wkh, Wvh, Wph);
  qkv_kernel<<<dim3(N_POS / 16, 2, 3), 256, 0, stream>>>(Hbuf, Wqh, Wkh, Wvh,
                                                         bq, bk, bv, Qbuf, Kbuf, Vbuf);
  attn_kernel<<<dim3(N_POS / 32, 2), 256, 0, stream>>>(Qbuf, Kbuf, Vbuf, Obuf);
  proj_kernel<<<dim3(N_POS / 16, 2), 256, 0, stream>>>(Obuf, Wph, bp, x, out);
}
